// MultiHeadAttention_46119358825232
// MI455X (gfx1250) — matrix operand order repaired and checked by reading, not compiled, not run on hardware
//
#include <hip/hip_runtime.h>
#include <math.h>

// ---------- types ----------
typedef __attribute__((ext_vector_type(16))) __bf16 v16bf;
typedef __attribute__((ext_vector_type(8)))  float  v8f;
typedef unsigned short bf16s;   // bf16 storage
typedef unsigned int   u32;

#define B_NUM 4
#define S_LEN 2048
#define D_DIM 1024
#define H_NUM 16
#define DK    64

__device__ inline bf16s f2bf(float f) {            // round-to-nearest-even
  u32 u = __float_as_uint(f);
  u32 r = (u + 0x7FFFu + ((u >> 16) & 1u)) >> 16;
  return (bf16s)r;
}
__device__ inline float bf2f(bf16s h) { return __uint_as_float(((u32)h) << 16); }
__device__ inline float ldx(const float* p) { return *p; }
__device__ inline float ldx(const bf16s* p) { return bf2f(*p); }

union Frag { v16bf v; u32 u[8]; };

__device__ inline v8f v8zero() {
  v8f z = {0.f, 0.f, 0.f, 0.f, 0.f, 0.f, 0.f, 0.f};
  return z;
}

__device__ inline v8f wmma_bf16(const Frag& a, const Frag& b, v8f c) {
  // D(f32 16x16) = A(bf16 16x32) * B(bf16 32x16) + C
  return __builtin_amdgcn_wmma_f32_16x16x32_bf16(false, a.v, false, b.v,
                                                 (short)0, c, false, false);
}

// ---------- mask bit-pack: int32[B,1,S,S] -> u32 bits [B,S,S/32] ----------
__global__ void __launch_bounds__(256)
maskpack_kernel(const int* __restrict__ mask, u32* __restrict__ mp, int nwords) {
  int idx = blockIdx.x * 256 + threadIdx.x;
  if (idx >= nwords) return;
  const int* src = mask + (long)idx * 32;
  u32 wd = 0;
  #pragma unroll
  for (int i = 0; i < 32; ++i) wd |= (src[i] != 0 ? 1u : 0u) << i;
  mp[idx] = wd;
}

// ---------- Y = (X @ W^T + bias) * scale ----------
// MODE 0: bf16 out, head-split [B,H,S,DK]      (Q, K)
// MODE 1: bf16 out, head-split T [B,H,DK,S]    (V)
// MODE 2: f32  out, row-major [M,N]            (final projection)
template <typename XT, int MODE>
__global__ void __launch_bounds__(256)
gemm_wt_kernel(const XT* __restrict__ X, const float* __restrict__ W,
               const float* __restrict__ bias, void* __restrict__ Yv, float scale) {
  const int n0 = blockIdx.x * 128;
  const int m0 = blockIdx.y * 128;
  const int tid = threadIdx.x;
  const int lane = tid & 31, wv = tid >> 5;
  const int lh = lane >> 4, ln = lane & 15;
  const int wm = wv >> 2, wn = wv & 3;   // wave grid 2 x 4

  __shared__ bf16s Xs[128][40];  // padded stride
  __shared__ bf16s Ws[128][40];

  v8f acc[4][2];
  #pragma unroll
  for (int i = 0; i < 4; ++i)
    #pragma unroll
    for (int j = 0; j < 2; ++j) acc[i][j] = v8zero();

  const int row = tid >> 1, c0 = (tid & 1) * 16;

  for (int kk = 0; kk < D_DIM; kk += 32) {
    __syncthreads();
    {
      const XT* xs = X + (long)(m0 + row) * D_DIM + kk + c0;
      #pragma unroll
      for (int i = 0; i < 16; ++i) Xs[row][c0 + i] = f2bf(ldx(xs + i));
      const float* ws = W + (long)(n0 + row) * D_DIM + kk + c0;
      #pragma unroll
      for (int i = 0; i < 16; ++i) Ws[row][c0 + i] = f2bf(ws[i]);
    }
    __syncthreads();

    Frag a[4], bq[2];
    #pragma unroll
    for (int i = 0; i < 4; ++i)
      #pragma unroll
      for (int v = 0; v < 8; ++v) {           // A-layout 16x32 bf16
        int k0 = (v >> 2) * 16 + lh * 8 + (v & 3) * 2;
        a[i].u[v] = *(const u32*)&Xs[wm * 64 + i * 16 + ln][k0];
      }
    #pragma unroll
    for (int j = 0; j < 2; ++j)
      #pragma unroll
      for (int v = 0; v < 8; ++v) {           // B-layout 32x16: B[k][n]=W[n][k]
        int k0 = (v >> 2) * 16 + lh * 8 + (v & 3) * 2;
        bq[j].u[v] = *(const u32*)&Ws[wn * 32 + j * 16 + ln][k0];
      }
    #pragma unroll
    for (int i = 0; i < 4; ++i)
      #pragma unroll
      for (int j = 0; j < 2; ++j) acc[i][j] = wmma_bf16(a[i], bq[j], acc[i][j]);
  }

  #pragma unroll
  for (int j = 0; j < 2; ++j) {
    const int n = n0 + wn * 32 + j * 16 + ln;
    const float bvv = bias[n];
    #pragma unroll
    for (int i = 0; i < 4; ++i) {
      #pragma unroll
      for (int r = 0; r < 8; ++r) {
        const int m = m0 + wm * 64 + i * 16 + r + 8 * lh;   // C/D layout
        float val = (acc[i][j][r] + bvv) * scale;
        if (MODE == 2) {
          ((float*)Yv)[(long)m * D_DIM + n] = val;
        } else {
          const int bb = m >> 11, s = m & (S_LEN - 1);
          const int hh = n >> 6, dk = n & 63;
          bf16s* Y = (bf16s*)Yv;
          if (MODE == 0)
            Y[(((long)bb * H_NUM + hh) * S_LEN + s) * DK + dk] = f2bf(val);
          else
            Y[(((long)bb * H_NUM + hh) * DK + dk) * S_LEN + s] = f2bf(val);
        }
      }
    }
  }
}

// ---------- flash attention: softmax(mask(Q K^T)) V  ----------
// grid: (S/64, H, B); 128 threads = 4 waves; each wave owns 16 query rows.
__global__ void __launch_bounds__(128)
flash_attn_kernel(const bf16s* __restrict__ Qh, const bf16s* __restrict__ Kh,
                  const bf16s* __restrict__ Vt, const u32* __restrict__ mp,
                  bf16s* __restrict__ attnO) {
  const int qb = blockIdx.x, h = blockIdx.y, b = blockIdx.z;
  const int tid = threadIdx.x, lane = tid & 31, wv = tid >> 5;
  const int lh = lane >> 4, ln = lane & 15;

  __shared__ bf16s Ks[64][72];       // [key][dk]
  __shared__ bf16s Vs[64][72];       // [dk][key]  (V pre-transposed in global)
  __shared__ bf16s Ps[4][16][72];    // per-wave P staging [qrow][key]
  __shared__ u32   Ms[4][32];        // per-wave mask words

  const long bh = (long)b * H_NUM + h;
  const bf16s* Qb = Qh + (bh * S_LEN + (long)qb * 64) * DK;
  const bf16s* Kb = Kh + bh * S_LEN * DK;
  const bf16s* Vb = Vt + bh * (long)DK * S_LEN;
  const int q0 = qb * 64;

  // Q fragments (K-dim = DK = 64 -> two 32-steps), scale already folded in
  Frag qf[2];
  #pragma unroll
  for (int st = 0; st < 2; ++st)
    #pragma unroll
    for (int v = 0; v < 8; ++v) {
      int k0 = st * 32 + (v >> 2) * 16 + lh * 8 + (v & 3) * 2;
      qf[st].u[v] = *(const u32*)(Qb + (long)(wv * 16 + ln) * DK + k0);
    }

  v8f o[4];
  #pragma unroll
  for (int nt = 0; nt < 4; ++nt) o[nt] = v8zero();
  float mr[8], lr[8];
  #pragma unroll
  for (int r = 0; r < 8; ++r) { mr[r] = -INFINITY; lr[r] = 0.f; }

  for (int kb = 0; kb < S_LEN; kb += 64) {
    __syncthreads();
    {
      const int rr = tid >> 1, cc = (tid & 1) * 32;
      const u32* sk = (const u32*)(Kb + (long)(kb + rr) * DK + cc);
      u32* dk_ = (u32*)&Ks[rr][cc];
      #pragma unroll
      for (int i = 0; i < 16; ++i) dk_[i] = sk[i];
      const u32* sv = (const u32*)(Vb + (long)rr * S_LEN + kb + cc);
      u32* dv = (u32*)&Vs[rr][cc];
      #pragma unroll
      for (int i = 0; i < 16; ++i) dv[i] = sv[i];
    }
    {
      const int rrow = lane >> 1, wsel = lane & 1;
      Ms[wv][lane] =
          mp[((long)b * S_LEN + q0 + wv * 16 + rrow) * (S_LEN / 32) + (kb >> 5) + wsel];
    }
    __syncthreads();

    // scores: 16 q-rows x 64 keys = 4 tiles
    v8f sc[4];
    #pragma unroll
    for (int kt = 0; kt < 4; ++kt) {
      v8f c = v8zero();
      #pragma unroll
      for (int st = 0; st < 2; ++st) {
        Frag bf;
        #pragma unroll
        for (int v = 0; v < 8; ++v) {      // B[d][key] = Ks[key][d]
          int d0 = st * 32 + (v >> 2) * 16 + lh * 8 + (v & 3) * 2;
          bf.u[v] = *(const u32*)&Ks[kt * 16 + ln][d0];
        }
        c = wmma_bf16(qf[st], bf, c);
      }
      sc[kt] = c;
    }

    // mask (same -1e9 semantics as reference)
    #pragma unroll
    for (int kt = 0; kt < 4; ++kt)
      #pragma unroll
      for (int r = 0; r < 8; ++r) {
        u32 mw = Ms[wv][(r + 8 * lh) * 2 + (kt >> 1)];
        if (!((mw >> ((kt & 1) * 16 + ln)) & 1u)) sc[kt][r] = -1e9f;
      }

    // online softmax (row m = r + 8*lh lives in a 16-lane group)
    float al[8];
    #pragma unroll
    for (int r = 0; r < 8; ++r) {
      float v0 = fmaxf(fmaxf(sc[0][r], sc[1][r]), fmaxf(sc[2][r], sc[3][r]));
      #pragma unroll
      for (int mk = 1; mk < 16; mk <<= 1) v0 = fmaxf(v0, __shfl_xor(v0, mk, 32));
      float mnew = fmaxf(mr[r], v0);
      al[r] = __expf(mr[r] - mnew);
      mr[r] = mnew;
    }
    float rsum[8];
    #pragma unroll
    for (int r = 0; r < 8; ++r) rsum[r] = 0.f;
    #pragma unroll
    for (int kt = 0; kt < 4; ++kt)
      #pragma unroll
      for (int r = 0; r < 8; ++r) {
        float p = __expf(sc[kt][r] - mr[r]);
        sc[kt][r] = p;
        rsum[r] += p;
      }
    #pragma unroll
    for (int r = 0; r < 8; ++r) {
      float v0 = rsum[r];
      #pragma unroll
      for (int mk = 1; mk < 16; mk <<= 1) v0 += __shfl_xor(v0, mk, 32);
      lr[r] = lr[r] * al[r] + v0;
    }
    #pragma unroll
    for (int nt = 0; nt < 4; ++nt)
      #pragma unroll
      for (int r = 0; r < 8; ++r) o[nt][r] *= al[r];

    // D-layout P -> LDS -> A-layout (same-wave, DS ops stay in order)
    #pragma unroll
    for (int kt = 0; kt < 4; ++kt)
      #pragma unroll
      for (int r = 0; r < 8; ++r)
        Ps[wv][r + 8 * lh][kt * 16 + ln] = f2bf(sc[kt][r]);

    Frag pf[2];
    #pragma unroll
    for (int st = 0; st < 2; ++st)
      #pragma unroll
      for (int v = 0; v < 8; ++v) {
        int k0 = st * 32 + (v >> 2) * 16 + lh * 8 + (v & 3) * 2;
        pf[st].u[v] = *(const u32*)&Ps[wv][ln][k0];
      }

    // O += P @ V   (B[k=key][n=dk] = Vs[dk][key], pairs contiguous along key)
    #pragma unroll
    for (int nt = 0; nt < 4; ++nt)
      #pragma unroll
      for (int st = 0; st < 2; ++st) {
        Frag vf;
        #pragma unroll
        for (int v = 0; v < 8; ++v) {
          int k0 = st * 32 + (v >> 2) * 16 + lh * 8 + (v & 3) * 2;
          vf.u[v] = *(const u32*)&Vs[nt * 16 + ln][k0];
        }
        o[nt] = wmma_bf16(pf[st], vf, o[nt]);
      }
  }

  // write O/l as bf16 into [B,S,D] for the final projection
  #pragma unroll
  for (int nt = 0; nt < 4; ++nt)
    #pragma unroll
    for (int r = 0; r < 8; ++r) {
      float val = o[nt][r] / lr[r];
      int qrow = q0 + wv * 16 + r + 8 * lh;
      attnO[((long)b * S_LEN + qrow) * D_DIM + h * DK + nt * 16 + ln] = f2bf(val);
    }
}

// ---------- launch ----------
extern "C" void kernel_launch(void* const* d_in, const int* in_sizes, int n_in,
                              void* d_out, int out_size, void* d_ws, size_t ws_size,
                              hipStream_t stream) {
  (void)in_sizes; (void)n_in; (void)out_size; (void)ws_size;
  const float* q    = (const float*)d_in[0];
  const float* k    = (const float*)d_in[1];
  const float* v    = (const float*)d_in[2];
  const int*   mask = (const int*)d_in[3];
  const float* wq   = (const float*)d_in[4];
  const float* bq   = (const float*)d_in[5];
  const float* wk   = (const float*)d_in[6];
  const float* bk   = (const float*)d_in[7];
  const float* wvw  = (const float*)d_in[8];
  const float* bv   = (const float*)d_in[9];
  const float* wo   = (const float*)d_in[10];
  const float* bo   = (const float*)d_in[11];

  char* ws = (char*)d_ws;
  const size_t bhsd = (size_t)B_NUM * H_NUM * S_LEN * DK;   // 8,388,608 elems
  bf16s* Qh = (bf16s*)ws; ws += bhsd * 2;
  bf16s* Kh = (bf16s*)ws; ws += bhsd * 2;
  bf16s* Vt = (bf16s*)ws; ws += bhsd * 2;
  bf16s* aO = (bf16s*)ws; ws += bhsd * 2;
  u32*  mpk = (u32*)ws;                                     // 2 MB

  const int nwords = B_NUM * S_LEN * (S_LEN / 32);
  maskpack_kernel<<<dim3((nwords + 255) / 256), 256, 0, stream>>>(mask, mpk, nwords);

  dim3 ggrid(D_DIM / 128, (B_NUM * S_LEN) / 128);
  gemm_wt_kernel<float, 0><<<ggrid, 256, 0, stream>>>(q, wq, bq, Qh, 0.125f);  // 1/sqrt(64)
  gemm_wt_kernel<float, 0><<<ggrid, 256, 0, stream>>>(k, wk, bk, Kh, 1.0f);
  gemm_wt_kernel<float, 1><<<ggrid, 256, 0, stream>>>(v, wvw, bv, Vt, 1.0f);

  flash_attn_kernel<<<dim3(S_LEN / 64, H_NUM, B_NUM), 128, 0, stream>>>(Qh, Kh, Vt, mpk, aO);

  gemm_wt_kernel<bf16s, 2><<<ggrid, 256, 0, stream>>>(aO, wo, bo, (float*)d_out, 1.0f);
}
